// ConvolutionEncoder_PWS_72421738545915
// MI455X (gfx1250) — matrix operand order repaired and checked by reading, not compiled, not run on hardware
//
#include <hip/hip_runtime.h>
#include <hip/hip_bf16.h>
#include <math.h>

// ---------------------------------------------------------------------------
// Problem constants (from the reference)
// ---------------------------------------------------------------------------
constexpr int Bn  = 4096;          // batch
constexpr int Ls  = 28;            // sequence length
constexpr int Vv  = 32000;         // vocab
constexpr int Dd  = 300;           // embedding dim
constexpr int K1  = Dd * Ls;       // 8400  (gram-1 reduction dim)
constexpr int K1p = 8448;          // padded to multiple of 32
constexpr int C1  = 300, T1 = 13;  // conv1 out
constexpr int C2  = 600, T2 = 5;   // conv2 out
constexpr int C3  = 100;           // conv3 out (T=1)
constexpr int KHp = 128;           // hidden gram K (100 -> 128 padded)

#define EPS_BN     1e-5f
#define EPS_RENORM 1e-7f

typedef __attribute__((ext_vector_type(16))) _Float16 v16h;
typedef __attribute__((ext_vector_type(8)))  _Float16 v8h;
typedef __attribute__((ext_vector_type(8)))  float    v8f;

// ---------------------------------------------------------------------------
// 1) Per-vocab-row max_norm=1 renorm scale: min(1, 1/(||row|| + eps))
// ---------------------------------------------------------------------------
__global__ void row_scale_kernel(const float* __restrict__ emb,
                                 float* __restrict__ scale) {
  int row  = blockIdx.x * 8 + (threadIdx.x >> 5);
  int lane = threadIdx.x & 31;
  if (row >= Vv) return;
  const float* r = emb + (size_t)row * Dd;
  float s = 0.f;
  for (int d = lane; d < Dd; d += 32) { float v = r[d]; s += v * v; }
  #pragma unroll
  for (int off = 16; off > 0; off >>= 1) s += __shfl_xor(s, off, 32);
  if (lane == 0) {
    float n = sqrtf(s);
    scale[row] = fminf(1.0f, 1.0f / (n + EPS_RENORM));
  }
}

// ---------------------------------------------------------------------------
// 2) Gather embedding rows -> E32 (fp32 conv1 input) + E16 (f16 gram matrix)
// ---------------------------------------------------------------------------
__global__ void gather_kernel(const int* __restrict__ x,
                              const float* __restrict__ emb,
                              const float* __restrict__ scale,
                              float* __restrict__ E32,
                              _Float16* __restrict__ E16) {
  int tok = blockIdx.x;                // b*Ls + l
  int b = tok / Ls, l = tok - b * Ls;
  int idx = x[tok];
  float s = scale[idx];
  const float* src = emb + (size_t)idx * Dd;
  float*     d32 = E32 + (size_t)b * K1  + l;
  _Float16*  d16 = E16 + (size_t)b * K1p + l;
  for (int d = threadIdx.x; d < Dd; d += blockDim.x) {
    float v = src[d] * s;
    d32[d * Ls] = v;
    d16[d * Ls] = (_Float16)v;
  }
}

// zero the f16 pad region [K1, K1p) of every row
__global__ void pad16_kernel(_Float16* __restrict__ E16) {
  int i = blockIdx.x * blockDim.x + threadIdx.x;   // over Bn*(K1p-K1)
  int b = i / (K1p - K1), r = i - b * (K1p - K1);
  if (b < Bn) E16[(size_t)b * K1p + K1 + r] = (_Float16)0.f;
}

// ---------------------------------------------------------------------------
// 3) conv1: [B,300,28] -> [B,300,13], k=5, stride=2, pad=1
// ---------------------------------------------------------------------------
__global__ void conv1_kernel(const float* __restrict__ E32,
                             const float* __restrict__ w1,
                             const float* __restrict__ b1,
                             float* __restrict__ y1) {
  __shared__ float xs[C1 * Ls];
  int b = blockIdx.x;
  const float* xin = E32 + (size_t)b * K1;
  for (int i = threadIdx.x; i < C1 * Ls; i += blockDim.x) xs[i] = xin[i];
  __syncthreads();
  for (int o = threadIdx.x; o < C1 * T1; o += blockDim.x) {
    int co = o / T1, t = o - co * T1;
    const float* w = w1 + (size_t)co * (C1 * 5);
    float acc = b1[co];
    int p0 = 2 * t - 1;
    for (int ci = 0; ci < C1; ci++) {
      const float* xr = xs + ci * Ls;
      const float* wr = w + ci * 5;
      #pragma unroll
      for (int k = 0; k < 5; k++) {
        int p = p0 + k;
        float xv = (p >= 0) ? xr[p] : 0.f;    // p <= 27 always
        acc = fmaf(xv, wr[k], acc);
      }
    }
    y1[(size_t)b * (C1 * T1) + o] = acc;
  }
}

// ---------------------------------------------------------------------------
// conv2: [B,300,13] -> [B,600,5], k=5, stride=2, pad=0
// ---------------------------------------------------------------------------
__global__ void conv2_kernel(const float* __restrict__ h1,
                             const float* __restrict__ w2,
                             const float* __restrict__ b2,
                             float* __restrict__ y2) {
  __shared__ float xs[C1 * T1];
  int b = blockIdx.x;
  const float* xin = h1 + (size_t)b * (C1 * T1);
  for (int i = threadIdx.x; i < C1 * T1; i += blockDim.x) xs[i] = xin[i];
  __syncthreads();
  for (int o = threadIdx.x; o < C2 * T2; o += blockDim.x) {
    int co = o / T2, t = o - co * T2;
    const float* w = w2 + (size_t)co * (C1 * 5);
    float acc = b2[co];
    for (int ci = 0; ci < C1; ci++) {
      const float* xr = xs + ci * T1 + 2 * t;
      const float* wr = w + ci * 5;
      #pragma unroll
      for (int k = 0; k < 5; k++) acc = fmaf(xr[k], wr[k], acc);
    }
    y2[(size_t)b * (C2 * T2) + o] = acc;
  }
}

// ---------------------------------------------------------------------------
// 4) BatchNorm training stats: per channel mean & 1/sqrt(var+eps) over (B,T)
// ---------------------------------------------------------------------------
__global__ void bn_stats_kernel(const float* __restrict__ y, int C, int T,
                                float* __restrict__ mean,
                                float* __restrict__ istd) {
  int c = blockIdx.x;
  int n = Bn * T;
  float s = 0.f, s2 = 0.f;
  for (int e = threadIdx.x; e < n; e += blockDim.x) {
    int b = e / T, t = e - b * T;
    float v = y[(size_t)b * C * T + (size_t)c * T + t];
    s += v; s2 += v * v;
  }
  __shared__ float rs[16], rs2[16];
  int lane = threadIdx.x & 31, w = threadIdx.x >> 5;
  #pragma unroll
  for (int off = 16; off > 0; off >>= 1) {
    s  += __shfl_xor(s,  off, 32);
    s2 += __shfl_xor(s2, off, 32);
  }
  if (lane == 0) { rs[w] = s; rs2[w] = s2; }
  __syncthreads();
  if (w == 0) {
    int nw = blockDim.x >> 5;
    s  = (lane < nw) ? rs[lane]  : 0.f;
    s2 = (lane < nw) ? rs2[lane] : 0.f;
    #pragma unroll
    for (int off = 16; off > 0; off >>= 1) {
      s  += __shfl_xor(s,  off, 32);
      s2 += __shfl_xor(s2, off, 32);
    }
    if (lane == 0) {
      float m   = s / (float)n;
      float var = s2 / (float)n - m * m;
      mean[c] = m;
      istd[c] = rsqrtf(var + EPS_BN);
    }
  }
}

// normalize + affine + ReLU, in place
__global__ void bn_apply_relu_kernel(float* __restrict__ y, int C, int T,
                                     const float* __restrict__ mean,
                                     const float* __restrict__ istd,
                                     const float* __restrict__ g,
                                     const float* __restrict__ be,
                                     size_t total) {
  size_t i = (size_t)blockIdx.x * blockDim.x + threadIdx.x;
  if (i >= total) return;
  int c = (int)((i / (size_t)T) % (size_t)C);
  float v = (y[i] - mean[c]) * istd[c] * g[c] + be[c];
  y[i] = fmaxf(v, 0.f);
}

// ---------------------------------------------------------------------------
// 5) conv3 + tanh + row L2-normalize
// ---------------------------------------------------------------------------
__global__ void conv3_kernel(const float* __restrict__ h2,
                             const float* __restrict__ w3,
                             const float* __restrict__ b3,
                             float* __restrict__ outh,
                             _Float16* __restrict__ F16) {
  __shared__ float xs[C2 * T2];
  __shared__ float red[4];
  int b = blockIdx.x;
  const float* xin = h2 + (size_t)b * (C2 * T2);
  for (int i = threadIdx.x; i < C2 * T2; i += blockDim.x) xs[i] = xin[i];
  __syncthreads();
  float o = 0.f;
  int co = threadIdx.x;
  if (co < C3) {
    float acc = b3[co];
    const float* w = w3 + (size_t)co * (C2 * 5);
    for (int i = 0; i < C2 * 5; i++) acc = fmaf(xs[i], w[i], acc);
    o = tanhf(acc);
  }
  float s = o * o;
  int lane = threadIdx.x & 31, wv = threadIdx.x >> 5;
  #pragma unroll
  for (int off = 16; off > 0; off >>= 1) s += __shfl_xor(s, off, 32);
  if (lane == 0) red[wv] = s;
  __syncthreads();
  float inv = rsqrtf(red[0] + red[1] + red[2] + red[3]);
  if (co < C3) {
    outh[(size_t)b * C3 + co] = o;
    F16[(size_t)b * KHp + co] = (_Float16)(o * inv);
  } else {
    F16[(size_t)b * KHp + co] = (_Float16)0.f;   // pad K 100..127
  }
}

// ---------------------------------------------------------------------------
// 6) Gram via WMMA with async-DMA B staging (double-buffered LDS):
//    C = A * A^T, A = [4096, Kp] f16 row-major, Kp % 32 == 0.
//
//    Per k-step: 16 independent ds_load_b128 issued as one clause ->
//    single s_wait_dscnt -> 8 back-to-back v_wmma_f32_16x16x32_f16.
//    B tile filled by global_load_async_to_lds_b128 (ASYNCcnt),
//    A fragment register-pipelined one k-step ahead.
//    LDS: 2 * 128 * 40 * 2B = 20 KB (40-half column stride: 16B-aligned
//    chunks, conflict-free fragment reads).
// ---------------------------------------------------------------------------
constexpr int BST = 40;   // halves per staged column (32 data + 8 pad)

__global__ __launch_bounds__(256)
void gram_wmma_kernel(const _Float16* __restrict__ A, int Kp,
                      float* __restrict__ pws) {
  __shared__ _Float16 Bt[2][128 * BST];
  int ti = blockIdx.y, tj = blockIdx.x;
  if (tj < ti) return;                       // only upper-triangle tiles
  int tid  = threadIdx.x;
  int lane = tid & 31;
  int wave = tid >> 5;
  int m     = lane & 15;
  int aKoff = (lane >> 4) * 8;
  const _Float16* arow = A + (size_t)(ti * 128 + wave * 16 + m) * Kp + aKoff;

  // cooperative B-tile fill: 512 b128 chunks; this thread owns chunks
  // (tid) and (tid+256): chunk -> column c = chunk>>2, sub j = chunk&3
  int c0 = tid >> 2,         j0 = tid & 3;
  int c1 = (tid + 256) >> 2, j1 = (tid + 256) & 3;
  const _Float16* g0 = A + (size_t)(tj * 128 + c0) * Kp + j0 * 8;
  const _Float16* g1 = A + (size_t)(tj * 128 + c1) * Kp + j1 * 8;
  unsigned lds0 = (unsigned)(size_t)&Bt[0][c0 * BST + j0 * 8];
  unsigned lds1 = (unsigned)(size_t)&Bt[0][c1 * BST + j1 * 8];
  unsigned bufstep = (unsigned)((size_t)&Bt[1][0] - (size_t)&Bt[0][0]);

  v8f acc[8];
  #pragma unroll
  for (int nt = 0; nt < 8; nt++)
    #pragma unroll
    for (int e = 0; e < 8; e++) acc[nt][e] = 0.f;

  // prologue: async-fill buffer 0 with k-chunk 0
  {
    unsigned long long a0 = (unsigned long long)(size_t)g0;
    unsigned long long a1 = (unsigned long long)(size_t)g1;
    asm volatile("global_load_async_to_lds_b128 %0, %2, off\n\t"
                 "global_load_async_to_lds_b128 %1, %3, off"
                 :: "v"(lds0), "v"(lds1), "v"(a0), "v"(a1) : "memory");
    asm volatile("s_wait_asynccnt 0x0" ::: "memory");
  }
  __syncthreads();

  v8h alo = *(const v8h*)(arow);
  v8h ahi = *(const v8h*)(arow + 16);

  const _Float16* bfrag0 = &Bt[0][(lane & 15) * BST + (lane >> 4) * 8];

  int s = 0;
  for (int k0 = 0; k0 < Kp; k0 += 32) {
    int kn   = k0 + 32;
    bool more = kn < Kp;
    // issue async fill of the other buffer for the next k-chunk
    if (more) {
      unsigned nl0 = lds0 + (unsigned)(s ^ 1) * bufstep;
      unsigned nl1 = lds1 + (unsigned)(s ^ 1) * bufstep;
      unsigned long long a0 = (unsigned long long)(size_t)(g0 + kn);
      unsigned long long a1 = (unsigned long long)(size_t)(g1 + kn);
      asm volatile("global_load_async_to_lds_b128 %0, %2, off\n\t"
                   "global_load_async_to_lds_b128 %1, %3, off"
                   :: "v"(nl0), "v"(nl1), "v"(a0), "v"(a1) : "memory");
    }
    // prefetch next A fragment (clamped to a valid address on last iter)
    int ka = more ? kn : 0;
    v8h alo_n = *(const v8h*)(arow + ka);
    v8h ahi_n = *(const v8h*)(arow + ka + 16);

    v16h a;
    #pragma unroll
    for (int i = 0; i < 8; i++) { a[i] = alo[i]; a[8 + i] = ahi[i]; }

    // phase 1: issue ALL 16 B-fragment LDS loads (one clause, one dscnt wait)
    const _Float16* bf = bfrag0 + (size_t)s * (128 * BST);
    v8h bl[8], bh[8];
    #pragma unroll
    for (int nt = 0; nt < 8; nt++) {
      const _Float16* lb = bf + nt * 16 * BST;
      bl[nt] = *(const v8h*)lb;
      bh[nt] = *(const v8h*)(lb + 16);
    }
    // phase 2: 8 back-to-back WMMAs
    #pragma unroll
    for (int nt = 0; nt < 8; nt++) {
      v16h bb;
      #pragma unroll
      for (int i = 0; i < 8; i++) { bb[i] = bl[nt][i]; bb[8 + i] = bh[nt][i]; }
      acc[nt] = __builtin_amdgcn_wmma_f32_16x16x32_f16(
          false, a, false, bb, (short)0, acc[nt], false, false);
    }

    asm volatile("s_wait_asynccnt 0x0" ::: "memory");
    __syncthreads();
    alo = alo_n; ahi = ahi_n;
    s ^= 1;
  }

  // C/D layout: lanes 0-15 -> M = r, lanes 16-31 -> M = r+8; N = lane&15
  int ibase = ti * 128 + wave * 16 + ((lane >> 4) ? 8 : 0);
  int jcol  = tj * 128 + (lane & 15);
  #pragma unroll
  for (int nt = 0; nt < 8; nt++) {
    int j = jcol + nt * 16;
    #pragma unroll
    for (int r = 0; r < 8; r++) {
      int i = ibase + r;
      if (i < j) {
        size_t idx = (size_t)i * (size_t)(2 * Bn - 1 - i) / 2
                   + (size_t)(j - i - 1);
        __builtin_nontemporal_store(acc[nt][r], &pws[idx]);
      }
    }
  }
}

// ---------------------------------------------------------------------------
// Host-side orchestration
// ---------------------------------------------------------------------------
extern "C" void kernel_launch(void* const* d_in, const int* in_sizes, int n_in,
                              void* d_out, int out_size, void* d_ws, size_t ws_size,
                              hipStream_t stream) {
  const int*   x   = (const int*)  d_in[0];
  const float* emb = (const float*)d_in[1];
  const float* w1  = (const float*)d_in[2];
  const float* b1  = (const float*)d_in[3];
  const float* w2  = (const float*)d_in[4];
  const float* b2  = (const float*)d_in[5];
  const float* w3  = (const float*)d_in[6];
  const float* b3  = (const float*)d_in[7];
  const float* g1  = (const float*)d_in[8];
  const float* be1 = (const float*)d_in[9];
  const float* g2  = (const float*)d_in[10];
  const float* be2 = (const float*)d_in[11];

  float* out = (float*)d_out;
  float*    out_h   = out;                                   // [B,100]
  float*    out_in  = out + (size_t)Bn * C3;                 // 8386560 pairs
  float*    out_hid = out_in + ((size_t)Bn * (Bn - 1)) / 2;  // 8386560 pairs

  char* ws = (char*)d_ws;
  size_t off = 0;
  auto carve = [&](size_t bytes) {
    void* p = ws + off;
    off = (off + bytes + 255) & ~(size_t)255;
    return p;
  };
  float*    scale = (float*)    carve((size_t)Vv * 4);
  float*    E32   = (float*)    carve((size_t)Bn * K1 * 4);
  _Float16* E16   = (_Float16*) carve((size_t)Bn * K1p * 2);
  float*    y1    = (float*)    carve((size_t)Bn * C1 * T1 * 4);
  float*    y2    = (float*)    carve((size_t)Bn * C2 * T2 * 4);
  _Float16* F16   = (_Float16*) carve((size_t)Bn * KHp * 2);
  float*    mean1 = (float*)    carve(C1 * 4);
  float*    istd1 = (float*)    carve(C1 * 4);
  float*    mean2 = (float*)    carve(C2 * 4);
  float*    istd2 = (float*)    carve(C2 * 4);
  (void)ws_size; (void)n_in; (void)in_sizes; (void)out_size;

  row_scale_kernel<<<(Vv + 7) / 8, 256, 0, stream>>>(emb, scale);

  gather_kernel<<<Bn * Ls, 128, 0, stream>>>(x, emb, scale, E32, E16);
  pad16_kernel<<<(Bn * (K1p - K1) + 255) / 256, 256, 0, stream>>>(E16);

  conv1_kernel<<<Bn, 256, 0, stream>>>(E32, w1, b1, y1);
  bn_stats_kernel<<<C1, 256, 0, stream>>>(y1, C1, T1, mean1, istd1);
  {
    size_t tot = (size_t)Bn * C1 * T1;
    bn_apply_relu_kernel<<<(unsigned)((tot + 255) / 256), 256, 0, stream>>>(
        y1, C1, T1, mean1, istd1, g1, be1, tot);
  }

  conv2_kernel<<<Bn, 256, 0, stream>>>(y1, w2, b2, y2);
  bn_stats_kernel<<<C2, 256, 0, stream>>>(y2, C2, T2, mean2, istd2);
  {
    size_t tot = (size_t)Bn * C2 * T2;
    bn_apply_relu_kernel<<<(unsigned)((tot + 255) / 256), 256, 0, stream>>>(
        y2, C2, T2, mean2, istd2, g2, be2, tot);
  }

  conv3_kernel<<<Bn, 128, 0, stream>>>(y2, w3, b3, out_h, F16);

  dim3 gtiles(Bn / 128, Bn / 128);
  gram_wmma_kernel<<<gtiles, 256, 0, stream>>>(E16, K1p, out_in);   // raw dots
  gram_wmma_kernel<<<gtiles, 256, 0, stream>>>(F16, KHp, out_hid);  // cosine
}
